// GNNModule_89215060672584
// MI455X (gfx1250) — matrix operand order repaired and checked by reading, not compiled, not run on hardware
//
#include <hip/hip_runtime.h>

// ---------------------------------------------------------------------------
// GNN: h1 = relu(x@Ws0^T + (A@x)@Wn0^T); out = relu(h1@Ws1^T + (A@h1)@Wn1^T)
// N=8192, D_IN=128, D_HID=256, D_OUT=128.  A is {0,1} int32 (exact in f16).
// Strategy: f16 WMMA (v_wmma_f32_16x16x32_f16, f32 accum). Aggregation streams
// the 268MB adjacency exactly once per layer (HBM-bound, ~23us floor total);
// activations kept in both row-major and transposed f16 so every WMMA fragment
// load is a contiguous b128.
// ---------------------------------------------------------------------------

typedef _Float16 v16h __attribute__((ext_vector_type(16)));
typedef _Float16 v8h  __attribute__((ext_vector_type(8)));
typedef float    v8f  __attribute__((ext_vector_type(8)));

#define NN 8192   // number of graph nodes

__device__ __forceinline__ v8f wmma16(v16h a, v16h b, v8f c) {
  // D = A(16x32 f16) * B(32x16 f16) + C(16x16 f32)
  return __builtin_amdgcn_wmma_f32_16x16x32_f16(
      /*neg_a=*/false, a, /*neg_b=*/false, b,
      /*c_mod=*/(short)0, c, /*reuse_a=*/false, /*reuse_b=*/false);
}

// -------------------------- format conversions -----------------------------

__global__ void cvt_f32_to_f16(const float* __restrict__ src,
                               _Float16* __restrict__ dst, int n) {
  int i = blockIdx.x * blockDim.x + threadIdx.x;
  if (i < n) dst[i] = (_Float16)src[i];
}

// src [R][C] f32  ->  dstT [C][R] f16
__global__ void cvt_f32_to_f16_T(const float* __restrict__ src,
                                 _Float16* __restrict__ dstT, int R, int C) {
  int i = blockIdx.x * blockDim.x + threadIdx.x;
  if (i < R * C) {
    int r = i / C, c = i % C;
    dstT[(size_t)c * R + r] = (_Float16)src[i];
  }
}

// ------------------------------ aggregation --------------------------------
// Out[N][D] (f16, row-major) = A[N][N] (int32 {0,1}) @ H, with H supplied
// transposed: Ht[D][N] f16.  One wave computes a 16-row x 128-col strip.
// A-fragment (16x32 f16) built inline from 4 x b128 int32 loads per lane.

template <int D>
__global__ __launch_bounds__(256) void agg_kernel(
    const int* __restrict__ A, const _Float16* __restrict__ Ht,
    _Float16* __restrict__ Out) {
  constexpr int NT = 8;                 // 8 col tiles = 128 cols per wave
  constexpr int CH = D / 128;           // column halves per row tile
  const int lane = threadIdx.x & 31;
  const int gw   = blockIdx.x * 8 + (threadIdx.x >> 5);
  const int rowTile = gw / CH;
  const int colBase = (gw % CH) * 128;
  const int row0 = rowTile * 16;
  const int g = lane >> 4;              // half-wave group (K sub-block)
  const int m = lane & 15;

  v8f acc[NT] = {};
  const int* arow = A + (size_t)(row0 + m) * NN;
  const int kk_add = g * 16;

  for (int k0 = 0; k0 < NN; k0 += 32) {
    __builtin_prefetch(arow + k0 + 1024, 0, 0);  // stream A ahead
    // A fragment: lanes 0-15 hold K 0..15, lanes 16-31 K 16..31
    const int* ap = arow + k0 + g * 16;
    int4 A0 = *(const int4*)(ap);
    int4 A1 = *(const int4*)(ap + 4);
    int4 A2 = *(const int4*)(ap + 8);
    int4 A3 = *(const int4*)(ap + 12);
    v16h af;
    af[0]  = (_Float16)A0.x; af[1]  = (_Float16)A0.y;
    af[2]  = (_Float16)A0.z; af[3]  = (_Float16)A0.w;
    af[4]  = (_Float16)A1.x; af[5]  = (_Float16)A1.y;
    af[6]  = (_Float16)A1.z; af[7]  = (_Float16)A1.w;
    af[8]  = (_Float16)A2.x; af[9]  = (_Float16)A2.y;
    af[10] = (_Float16)A2.z; af[11] = (_Float16)A2.w;
    af[12] = (_Float16)A3.x; af[13] = (_Float16)A3.y;
    af[14] = (_Float16)A3.z; af[15] = (_Float16)A3.w;

    const int kk = k0 + kk_add;         // B: lanes 0-15 K 0..15, 16-31 K 16..31
#pragma unroll
    for (int t = 0; t < NT; ++t) {
      const _Float16* bp = Ht + (size_t)(colBase + t * 16 + m) * NN + kk;
      v16h bf = *(const v16h*)bp;       // contiguous 32B (transposed storage)
      acc[t] = wmma16(af, bf, acc[t]);
    }
  }

  // C/D layout: VGPR r, lanes 0-15 -> (M=r, N=lane); lanes 16-31 -> (M=r+8)
#pragma unroll
  for (int t = 0; t < NT; ++t) {
    const int col = colBase + t * 16 + m;
#pragma unroll
    for (int r = 0; r < 8; ++r) {
      const int row = row0 + r + g * 8;
      Out[(size_t)row * D + col] = (_Float16)acc[t][r];
    }
  }
}

// ------------------------------ projection ---------------------------------
// relu(H@Ws^T + Ng@Wn^T).  H,Ng row-major f16 [N][DIN]; Ws,Wn f16 [DOUT][DIN]
// (row-major weight == already "B-transposed" for out = h @ W^T).
// One wave computes 16 rows x 64 cols.  Optionally emits f16 row-major +
// f16 transposed copies (feeding next layer) or f32 (final output).

template <int DIN, int DOUT, bool F32OUT>
__global__ __launch_bounds__(256) void proj_kernel(
    const _Float16* __restrict__ H, const _Float16* __restrict__ Ng,
    const _Float16* __restrict__ Ws, const _Float16* __restrict__ Wn,
    _Float16* __restrict__ OutH, _Float16* __restrict__ OutT,
    float* __restrict__ OutF) {
  constexpr int NT = 4;                 // 64 cols per wave
  constexpr int CG = DOUT / 64;
  const int lane = threadIdx.x & 31;
  const int gw   = blockIdx.x * 8 + (threadIdx.x >> 5);
  const int rowTile = gw / CG;
  const int colBase = (gw % CG) * 64;
  const int row0 = rowTile * 16;
  const int g = lane >> 4;
  const int m = lane & 15;

  v8f acc[NT] = {};
  const _Float16* hrow = H  + (size_t)(row0 + m) * DIN;
  const _Float16* nrow = Ng + (size_t)(row0 + m) * DIN;

  for (int k0 = 0; k0 < DIN; k0 += 32) {
    const int ka = k0 + g * 16;
    v8h h0 = *(const v8h*)(hrow + ka);
    v8h h1 = *(const v8h*)(hrow + ka + 8);
    v16h afH = __builtin_shufflevector(h0, h1, 0, 1, 2, 3, 4, 5, 6, 7,
                                       8, 9, 10, 11, 12, 13, 14, 15);
    v8h n0 = *(const v8h*)(nrow + ka);
    v8h n1 = *(const v8h*)(nrow + ka + 8);
    v16h afN = __builtin_shufflevector(n0, n1, 0, 1, 2, 3, 4, 5, 6, 7,
                                       8, 9, 10, 11, 12, 13, 14, 15);
    const int kb = k0 + g * 16;
#pragma unroll
    for (int t = 0; t < NT; ++t) {
      const size_t wro = (size_t)(colBase + t * 16 + m) * DIN + kb;
      v16h bs = *(const v16h*)(Ws + wro);
      acc[t] = wmma16(afH, bs, acc[t]);
      v16h bn = *(const v16h*)(Wn + wro);
      acc[t] = wmma16(afN, bn, acc[t]);
    }
  }

#pragma unroll
  for (int t = 0; t < NT; ++t) {
    const int col = colBase + t * 16 + m;
#pragma unroll
    for (int r = 0; r < 8; ++r) {
      const int row = row0 + r + g * 8;
      float v = acc[t][r];
      v = v > 0.f ? v : 0.f;            // ReLU
      if (F32OUT) {
        OutF[(size_t)row * DOUT + col] = v;
      } else {
        _Float16 hv = (_Float16)v;
        OutH[(size_t)row * DOUT + col] = hv;       // row-major (A-operand use)
        OutT[(size_t)col * NN + row]   = hv;       // transposed (B-operand use)
      }
    }
  }
}

// ------------------------------- launcher ----------------------------------

extern "C" void kernel_launch(void* const* d_in, const int* in_sizes, int n_in,
                              void* d_out, int out_size, void* d_ws,
                              size_t ws_size, hipStream_t stream) {
  const float* x   = (const float*)d_in[0];   // [8192,128]
  const int*   adj = (const int*)d_in[1];     // [8192,8192]
  const float* ws0 = (const float*)d_in[2];   // [256,128]
  const float* wn0 = (const float*)d_in[3];   // [256,128]
  const float* ws1 = (const float*)d_in[4];   // [128,256]
  const float* wn1 = (const float*)d_in[5];   // [128,256]
  float* out = (float*)d_out;                 // [8192,128]

  char* w = (char*)d_ws;                      // ~18.25 MB used
  _Float16* x_h  = (_Float16*)(w);                       // [8192][128]  2MB
  _Float16* x_t  = (_Float16*)(w + (2u << 20));          // [128][8192]  2MB
  _Float16* n0b  = (_Float16*)(w + (4u << 20));          // [8192][128]  2MB
  _Float16* h1   = (_Float16*)(w + (6u << 20));          // [8192][256]  4MB
  _Float16* h1t  = (_Float16*)(w + (10u << 20));         // [256][8192]  4MB
  _Float16* n1b  = (_Float16*)(w + (14u << 20));         // [8192][256]  4MB
  _Float16* ws0h = (_Float16*)(w + (18u << 20));         // 4 x 64KB
  _Float16* wn0h = ws0h + 32768;
  _Float16* ws1h = wn0h + 32768;
  _Float16* wn1h = ws1h + 32768;

  // Pre-convert activations/weights to f16 (x also stored transposed).
  cvt_f32_to_f16<<<4096, 256, 0, stream>>>(x, x_h, NN * 128);
  cvt_f32_to_f16_T<<<4096, 256, 0, stream>>>(x, x_t, NN, 128);
  cvt_f32_to_f16<<<128, 256, 0, stream>>>(ws0, ws0h, 256 * 128);
  cvt_f32_to_f16<<<128, 256, 0, stream>>>(wn0, wn0h, 256 * 128);
  cvt_f32_to_f16<<<128, 256, 0, stream>>>(ws1, ws1h, 128 * 256);
  cvt_f32_to_f16<<<128, 256, 0, stream>>>(wn1, wn1h, 128 * 256);

  // Layer 1: neigh0 = A @ x ; h1 = relu(x@Ws0^T + neigh0@Wn0^T)
  // agg<128>: 512 row-tiles x 1 col-half = 512 waves -> 64 blocks of 8 waves
  agg_kernel<128><<<64, 256, 0, stream>>>(adj, x_t, n0b);
  // proj: 512 row-tiles x (256/64)=4 col-groups = 2048 waves -> 256 blocks
  proj_kernel<128, 256, false><<<256, 256, 0, stream>>>(
      x_h, n0b, ws0h, wn0h, h1, h1t, nullptr);

  // Layer 2: neigh1 = A @ h1 ; out = relu(h1@Ws1^T + neigh1@Wn1^T)
  // agg<256>: 512 x 2 = 1024 waves -> 128 blocks
  agg_kernel<256><<<128, 256, 0, stream>>>(adj, h1t, n1b);
  // proj: 512 x (128/64)=2 = 1024 waves -> 128 blocks
  proj_kernel<256, 128, true><<<128, 256, 0, stream>>>(
      h1, n1b, ws1h, wn1h, nullptr, nullptr, out);
}
